// HashGenerator_34694745817199
// MI455X (gfx1250) — matrix operand order repaired and checked by reading, not compiled, not run on hardware
//
#include <hip/hip_runtime.h>
#include <math.h>

typedef __attribute__((ext_vector_type(16))) __bf16 v16bf;
typedef __attribute__((ext_vector_type(8)))  float  v8f;

#define LVL   16
#define TBLN  16384
#define IMG   256
#define NPIX  (IMG*IMG)
#define NB    8
#define SDIM  256
#define ZDIM  512
#define HID   64
#define FDIM  32
#define GAINC 1.4142135623730951f

// workspace layout (bytes)
#define WS_SCALE 0        // 8*32 f32            (1024 B)
#define WS_W0    1024     // 8*4*256 dwords bf16 (32768 B)   layer0: 4 n-tiles, 1 k-step
#define WS_W1    33792    // 8*8*256 dwords      (65536 B)   layer1: 4 n-tiles x 2 k-steps
#define WS_W2    99328    // 8*2*256 dwords      (16384 B)   layer2: 1 n-tile  x 2 k-steps

__device__ __forceinline__ float lrelu(float x) {
    return GAINC * (x >= 0.0f ? x : 0.2f * x);
}
__device__ __forceinline__ unsigned short f2bf(float f) {
    unsigned x = __float_as_uint(f);
    unsigned r = x + 0x7FFFu + ((x >> 16) & 1u);   // round-to-nearest-even
    return (unsigned short)(r >> 16);
}

// ---------------------------------------------------------------------------
// Prep kernel (1 block, 256 threads): mapping MLP, table scales, and per-batch
// modulated weights packed in WMMA B-fragment order as bf16.
// B-frag layout (16-bit, KxN=32x16): lane ln holds column n=ln&15;
// lanes 0-15 cover K=0..7,16..23, lanes 16-31 cover K=8..15,24..31; dword d = K {k0, k0+1}, k0=16(d>>2)+8(ln>>4)+2(d&3).
// ---------------------------------------------------------------------------
__global__ __launch_bounds__(256) void hash_prep(
    const float* __restrict__ z,
    const float* __restrict__ mw0, const float* __restrict__ mb0,
    const float* __restrict__ mw1, const float* __restrict__ mb1,
    const float* __restrict__ mw2, const float* __restrict__ mb2,
    const float* __restrict__ w_mod, const float* __restrict__ b_mod,
    const float* __restrict__ a0w, const float* __restrict__ a0b, const float* __restrict__ w0,
    const float* __restrict__ a1w, const float* __restrict__ a1b, const float* __restrict__ w1,
    const float* __restrict__ a2w, const float* __restrict__ a2b, const float* __restrict__ w2,
    float* __restrict__ ws_scale,
    unsigned* __restrict__ W0p, unsigned* __restrict__ W1p, unsigned* __restrict__ W2p)
{
    __shared__ float sA[NB][SDIM];
    __shared__ float sB[NB][SDIM];
    __shared__ float st0[NB][FDIM];
    __shared__ float st1[NB][HID];
    __shared__ float st2[NB][HID];
    __shared__ float dm0[NB][HID];
    __shared__ float dm1[NB][HID];
    __shared__ float dm2[NB][16];

    const int t = threadIdx.x;

    // mapping layer 0: z (8x512) @ mw0 (512x256)
    {
        float acc[NB];
        #pragma unroll
        for (int b = 0; b < NB; ++b) acc[b] = mb0[t];
        for (int k = 0; k < ZDIM; ++k) {
            float w = mw0[k * SDIM + t];
            #pragma unroll
            for (int b = 0; b < NB; ++b) acc[b] += z[b * ZDIM + k] * w;
        }
        #pragma unroll
        for (int b = 0; b < NB; ++b) sA[b][t] = lrelu(acc[b]);
    }
    __syncthreads();
    // mapping layer 1
    {
        float acc[NB];
        #pragma unroll
        for (int b = 0; b < NB; ++b) acc[b] = mb1[t];
        for (int k = 0; k < SDIM; ++k) {
            float w = mw1[k * SDIM + t];
            #pragma unroll
            for (int b = 0; b < NB; ++b) acc[b] += sA[b][k] * w;
        }
        #pragma unroll
        for (int b = 0; b < NB; ++b) sB[b][t] = lrelu(acc[b]);
    }
    __syncthreads();
    // mapping layer 2 -> s2 in sA
    {
        float acc[NB];
        #pragma unroll
        for (int b = 0; b < NB; ++b) acc[b] = mb2[t];
        for (int k = 0; k < SDIM; ++k) {
            float w = mw2[k * SDIM + t];
            #pragma unroll
            for (int b = 0; b < NB; ++b) acc[b] += sB[b][k] * w;
        }
        #pragma unroll
        for (int b = 0; b < NB; ++b) sA[b][t] = lrelu(acc[b]);
    }
    __syncthreads();

    // table scale (1 + s@w_mod + b_mod) and style0, one (b,j<32) per thread
    {
        int b = t >> 5, j = t & 31;
        float a = 1.0f + b_mod[j];
        float c0 = a0b[j];
        for (int k = 0; k < SDIM; ++k) {
            float s = sA[b][k];
            a  += s * w_mod[k * (2 * LVL) + j];
            c0 += s * a0w[k * FDIM + j];
        }
        ws_scale[b * 32 + j] = a;
        st0[b][j] = c0;
    }
    // style1, style2: (b, j<64), two tasks per thread
    for (int q = 0; q < 2; ++q) {
        int g = t + q * 256;
        int b = g >> 6, j = g & 63;
        float c1 = a1b[j], c2 = a2b[j];
        for (int k = 0; k < SDIM; ++k) {
            float s = sA[b][k];
            c1 += s * a1w[k * HID + j];
            c2 += s * a2w[k * HID + j];
        }
        st1[b][j] = c1;
        st2[b][j] = c2;
    }
    __syncthreads();

    // demod factors: rsqrt(style^2 @ w^2 + 1e-8)
    for (int q = 0; q < 2; ++q) {
        int g = t + q * 256;
        int b = g >> 6, j = g & 63;
        float d0 = 1e-8f, d1 = 1e-8f;
        for (int i = 0; i < FDIM; ++i) { float ww = w0[i * HID + j], ss = st0[b][i]; d0 += ss * ss * ww * ww; }
        for (int i = 0; i < HID;  ++i) { float ww = w1[i * HID + j], ss = st1[b][i]; d1 += ss * ss * ww * ww; }
        dm0[b][j] = rsqrtf(d0);
        dm1[b][j] = rsqrtf(d1);
    }
    if (t < 128) {
        int b = t >> 4, j = t & 15;
        float r = 0.0f;
        if (j < 3) {
            float d = 1e-8f;
            for (int i = 0; i < HID; ++i) { float ww = w2[i * 3 + j], ss = st2[b][i]; d += ss * ss * ww * ww; }
            r = rsqrtf(d);
        }
        dm2[b][j] = r;
    }
    __syncthreads();

    // Pack modulated weights in B-fragment order.
    // W0: 8 batches * 4 frags * 256 dwords
    for (int q = 0; q < 32; ++q) {
        int g = t + q * 256;
        int d = g & 7, ln = (g >> 3) & 31, nt = (g >> 8) & 3, b = g >> 10;
        int n  = nt * 16 + (ln & 15);
        int k0 = ((d >> 2) * 16) + ((ln >> 4) * 8) + 2 * (d & 3);
        float v0 = st0[b][k0]     * w0[k0 * HID + n]       * dm0[b][n];
        float v1 = st0[b][k0 + 1] * w0[(k0 + 1) * HID + n] * dm0[b][n];
        W0p[g] = (unsigned)f2bf(v0) | ((unsigned)f2bf(v1) << 16);
    }
    // W1: 8 batches * 8 frags (nt*2+ks) * 256 dwords
    for (int q = 0; q < 64; ++q) {
        int g = t + q * 256;
        int d = g & 7, ln = (g >> 3) & 31, frag = (g >> 8) & 7, b = g >> 11;
        int nt = frag >> 1, ks = frag & 1;
        int n  = nt * 16 + (ln & 15);
        int k0 = ks * 32 + ((d >> 2) * 16) + ((ln >> 4) * 8) + 2 * (d & 3);
        float v0 = st1[b][k0]     * w1[k0 * HID + n]       * dm1[b][n];
        float v1 = st1[b][k0 + 1] * w1[(k0 + 1) * HID + n] * dm1[b][n];
        W1p[g] = (unsigned)f2bf(v0) | ((unsigned)f2bf(v1) << 16);
    }
    // W2: 8 batches * 2 frags (ks) * 256 dwords; columns >=3 are zero
    for (int q = 0; q < 16; ++q) {
        int g = t + q * 256;
        int d = g & 7, ln = (g >> 3) & 31, ks = (g >> 8) & 1, b = g >> 9;
        int n  = ln & 15;
        int k0 = ks * 32 + ((d >> 2) * 16) + ((ln >> 4) * 8) + 2 * (d & 3);
        float v0 = 0.0f, v1 = 0.0f;
        if (n < 3) {
            v0 = st2[b][k0]     * w2[k0 * 3 + n]       * dm2[b][n];
            v1 = st2[b][k0 + 1] * w2[(k0 + 1) * 3 + n] * dm2[b][n];
        }
        W2p[g] = (unsigned)f2bf(v0) | ((unsigned)f2bf(v1) << 16);
    }
}

// ---------------------------------------------------------------------------
// Main kernel: one block = (batch, 128 pixels). Gather hash features -> LDS,
// then 3-layer WMMA bf16 MLP. Each wave owns 32 rows (2 M-tiles of 16).
// A-frag (16-bit 16x32): lane ln -> row m=ln&15; lanes 0-15: K 0-7 & 16-23,
// lanes 16-31: K 8-15 & 24-31 (2 K per dword).
// C/D layout: VGPR j -> M = j + 8*(ln>=16), N = ln&15.
// ---------------------------------------------------------------------------
__global__ __launch_bounds__(128) void hash_main(
    const float*    __restrict__ tbl,     // (L, T, 2) base tables
    const float*    __restrict__ scl,     // (B, 32)   per-batch table scales
    const unsigned* __restrict__ W0p,
    const unsigned* __restrict__ W1p,
    const unsigned* __restrict__ W2p,
    const float*    __restrict__ bb0,
    const float*    __restrict__ bb1,
    const float*    __restrict__ bb2,
    float*          __restrict__ out)     // (B, 3, IMG, IMG)
{
    __shared__ __bf16 hA[128][66];   // features (cols 0..31), later layer1 output (cols 0..63)
    __shared__ __bf16 hB[128][66];   // layer0 output

    const int b        = blockIdx.y;
    const int tileBase = blockIdx.x * 128;
    const int tid      = threadIdx.x;

    // ---- feature gather: one pixel per thread ----
    {
        const int p = tileBase + tid;
        const float xc = ((float)(p >> 8)  + 0.5f) * (1.0f / IMG);
        const float yc = ((float)(p & 255) + 0.5f) * (1.0f / IMG);
        const float res[LVL] = {16.f,19.f,23.f,27.f,33.f,40.f,48.f,58.f,
                                70.f,84.f,101.f,122.f,147.f,176.f,212.f,256.f};
        const float* sc = scl + b * 32;
        #pragma unroll
        for (int l = 0; l < LVL; ++l) {
            float px = xc * res[l], py = yc * res[l];
            float fx = floorf(px), fy = floorf(py);
            float tx = px - fx,    ty = py - fy;
            unsigned ux = (unsigned)fx, uy = (unsigned)fy;
            float a0 = 0.0f, a1 = 0.0f;
            #pragma unroll
            for (int c = 0; c < 4; ++c) {
                unsigned cx  = ux + (c & 1);
                unsigned cy  = uy + (c >> 1);
                unsigned idx = (cx ^ (cy * 2654435761u)) & (TBLN - 1);
                float wx = (c & 1)  ? tx : 1.0f - tx;
                float wy = (c >> 1) ? ty : 1.0f - ty;
                float w  = wx * wy;
                const float* f = tbl + ((size_t)l * TBLN + idx) * 2;
                a0 += w * f[0];
                a1 += w * f[1];
            }
            hA[tid][2 * l]     = (__bf16)(a0 * sc[2 * l]);
            hA[tid][2 * l + 1] = (__bf16)(a1 * sc[2 * l + 1]);
        }
    }
    __syncthreads();

    const int wv  = tid >> 5, ln = tid & 31;
    const int nlo = ln & 15,  khi = ln >> 4;
    union AB { v16bf v; unsigned u[8]; uint4 q[2]; };

    // ---- layer 0: hA(:, :32) @ W0 (32x64) -> hB ----
    #pragma unroll
    for (int mt = 0; mt < 2; ++mt) {
        const int mbase = wv * 32 + mt * 16;
        const int arow  = mbase + nlo;
        AB a;
        #pragma unroll
        for (int d = 0; d < 4; ++d) {
            a.u[d]     = *(const unsigned*)&hA[arow][khi * 8 + 2 * d];
            a.u[4 + d] = *(const unsigned*)&hA[arow][16 + khi * 8 + 2 * d];
        }
        #pragma unroll
        for (int nt = 0; nt < 4; ++nt) {
            AB bf;
            const uint4* src = (const uint4*)(W0p + ((b * 4 + nt) * 256 + ln * 8));
            bf.q[0] = src[0]; bf.q[1] = src[1];
            float bias = bb0[nt * 16 + nlo];
            v8f c;
            #pragma unroll
            for (int jj = 0; jj < 8; ++jj) c[jj] = bias;
            c = __builtin_amdgcn_wmma_f32_16x16x32_bf16(false, a.v, false, bf.v,
                                                        (short)0, c, false, false);
            #pragma unroll
            for (int jj = 0; jj < 8; ++jj) {
                float x = c[jj];
                x = GAINC * (x >= 0.0f ? x : 0.2f * x);
                hB[mbase + khi * 8 + jj][nt * 16 + nlo] = (__bf16)x;
            }
        }
    }
    __syncthreads();

    // ---- layer 1: hB(:, :64) @ W1 (64x64) -> hA ----
    #pragma unroll
    for (int mt = 0; mt < 2; ++mt) {
        const int mbase = wv * 32 + mt * 16;
        const int arow  = mbase + nlo;
        v8f acc[4];
        #pragma unroll
        for (int nt = 0; nt < 4; ++nt) {
            float bias = bb1[nt * 16 + nlo];
            #pragma unroll
            for (int jj = 0; jj < 8; ++jj) acc[nt][jj] = bias;
        }
        #pragma unroll
        for (int ks = 0; ks < 2; ++ks) {
            AB a;
            #pragma unroll
            for (int d = 0; d < 4; ++d) {
                a.u[d]     = *(const unsigned*)&hB[arow][ks * 32 + khi * 8 + 2 * d];
                a.u[4 + d] = *(const unsigned*)&hB[arow][ks * 32 + 16 + khi * 8 + 2 * d];
            }
            #pragma unroll
            for (int nt = 0; nt < 4; ++nt) {
                AB bf;
                const uint4* src = (const uint4*)(W1p + ((b * 8 + nt * 2 + ks) * 256 + ln * 8));
                bf.q[0] = src[0]; bf.q[1] = src[1];
                acc[nt] = __builtin_amdgcn_wmma_f32_16x16x32_bf16(false, a.v, false, bf.v,
                                                                  (short)0, acc[nt], false, false);
            }
        }
        #pragma unroll
        for (int nt = 0; nt < 4; ++nt) {
            #pragma unroll
            for (int jj = 0; jj < 8; ++jj) {
                float x = acc[nt][jj];
                x = GAINC * (x >= 0.0f ? x : 0.2f * x);
                hA[mbase + khi * 8 + jj][nt * 16 + nlo] = (__bf16)x;
            }
        }
    }
    __syncthreads();

    // ---- layer 2: hA(:, :64) @ W2 (64x16, cols>=3 zero) -> tanh -> out ----
    #pragma unroll
    for (int mt = 0; mt < 2; ++mt) {
        const int mbase = wv * 32 + mt * 16;
        const int arow  = mbase + nlo;
        int   bi   = (nlo < 3) ? nlo : 0;
        float bias = (nlo < 3) ? bb2[bi] : 0.0f;
        v8f c;
        #pragma unroll
        for (int jj = 0; jj < 8; ++jj) c[jj] = bias;
        #pragma unroll
        for (int ks = 0; ks < 2; ++ks) {
            AB a;
            #pragma unroll
            for (int d = 0; d < 4; ++d) {
                a.u[d]     = *(const unsigned*)&hA[arow][ks * 32 + khi * 8 + 2 * d];
                a.u[4 + d] = *(const unsigned*)&hA[arow][ks * 32 + 16 + khi * 8 + 2 * d];
            }
            AB bf;
            const uint4* src = (const uint4*)(W2p + ((b * 2 + ks) * 256 + ln * 8));
            bf.q[0] = src[0]; bf.q[1] = src[1];
            c = __builtin_amdgcn_wmma_f32_16x16x32_bf16(false, a.v, false, bf.v,
                                                        (short)0, c, false, false);
        }
        if (nlo < 3) {
            #pragma unroll
            for (int jj = 0; jj < 8; ++jj) {
                float x  = tanhf(c[jj]);
                int  row = mbase + khi * 8 + jj;
                out[(size_t)(b * 3 + nlo) * NPIX + tileBase + row] = x;
            }
        }
    }
}

extern "C" void kernel_launch(void* const* d_in, const int* in_sizes, int n_in,
                              void* d_out, int out_size, void* d_ws, size_t ws_size,
                              hipStream_t stream) {
    (void)in_sizes; (void)n_in; (void)out_size; (void)ws_size;
    const float* z     = (const float*)d_in[0];
    const float* mw0   = (const float*)d_in[1];
    const float* mb0   = (const float*)d_in[2];
    const float* mw1   = (const float*)d_in[3];
    const float* mb1   = (const float*)d_in[4];
    const float* mw2   = (const float*)d_in[5];
    const float* mb2   = (const float*)d_in[6];
    const float* tbl   = (const float*)d_in[7];
    const float* w_mod = (const float*)d_in[8];
    const float* b_mod = (const float*)d_in[9];
    const float* a0w   = (const float*)d_in[10];
    const float* a0b   = (const float*)d_in[11];
    const float* w0    = (const float*)d_in[12];
    const float* bb0   = (const float*)d_in[13];
    const float* a1w   = (const float*)d_in[14];
    const float* a1b   = (const float*)d_in[15];
    const float* w1    = (const float*)d_in[16];
    const float* bb1   = (const float*)d_in[17];
    const float* a2w   = (const float*)d_in[18];
    const float* a2b   = (const float*)d_in[19];
    const float* w2    = (const float*)d_in[20];
    const float* bb2   = (const float*)d_in[21];

    char* ws = (char*)d_ws;
    float*    wsScale = (float*)(ws + WS_SCALE);
    unsigned* W0p     = (unsigned*)(ws + WS_W0);
    unsigned* W1p     = (unsigned*)(ws + WS_W1);
    unsigned* W2p     = (unsigned*)(ws + WS_W2);

    hash_prep<<<1, 256, 0, stream>>>(z, mw0, mb0, mw1, mb1, mw2, mb2,
                                     w_mod, b_mod,
                                     a0w, a0b, w0,
                                     a1w, a1b, w1,
                                     a2w, a2b, w2,
                                     wsScale, W0p, W1p, W2p);

    hash_main<<<dim3(NPIX / 128, NB), 128, 0, stream>>>(
        tbl, wsScale, W0p, W1p, W2p, bb0, bb1, bb2, (float*)d_out);
}
